// BatchMultiHeadGraphAttention_21045339750533
// MI455X (gfx1250) — matrix operand order repaired and checked by reading, not compiled, not run on hardware
//
#include <hip/hip_runtime.h>
#include <hip/hip_bf16.h>

typedef __attribute__((ext_vector_type(16))) _Float16 v16h;
typedef __attribute__((ext_vector_type(8)))  _Float16 v8h;
typedef __attribute__((ext_vector_type(8)))  float    v8f;

constexpr int kBS   = 8;
constexpr int kN    = 1024;
constexpr int kNH   = 8;
constexpr int kFIN  = 256;
constexpr int kFOUT = 64;

constexpr int kWPitch = 264;  // halves; 528B rows: 16B aligned, bank-staggered
constexpr int kVPitch = 136;  // halves; 272B rows: 16B aligned, bank-staggered

constexpr float kLog2e = 1.4426950408889634f;

__device__ __forceinline__ float fast_tanh(float x) {
#if __has_builtin(__builtin_amdgcn_tanhf)
    return __builtin_amdgcn_tanhf(x);          // gfx1250 v_tanh_f32
#else
    const float e = __expf(2.f * x);           // branch-free fallback
    return 1.f - 2.f / (e + 1.f);
#endif
}

__device__ __forceinline__ float fast_exp2(float x) {
#if __has_builtin(__builtin_amdgcn_exp2f)
    return __builtin_amdgcn_exp2f(x);          // raw v_exp_f32
#else
    return exp2f(x);
#endif
}

// ---------------------------------------------------------------------------
// Phase 1: h_prime[b,h] = h[b] @ w[h] via f16 WMMA (f32 accum).
// Writes TRANSPOSED f16 h_prime [bh][col(64)][row(1024)] to workspace
// (so phase 2 B-fragments are contiguous), plus tanh-dot attention scalars.
// ---------------------------------------------------------------------------
__global__ __launch_bounds__(256) void gat_phase1(
    const float* __restrict__ h, const float* __restrict__ w,
    const float* __restrict__ a_src, const float* __restrict__ a_dst,
    _Float16* __restrict__ hp16T, float* __restrict__ as_ws, float* __restrict__ ad_ws)
{
    __shared__ _Float16 w16T[kFOUT * kWPitch];   // W transposed [n][k], ~33 KB

    const int bh   = blockIdx.x;
    const int b    = bh >> 3;
    const int hh   = bh & 7;
    const int tid  = threadIdx.x;
    const int wave = tid >> 5;
    const int lane = tid & 31;
    const int half = lane >> 4;
    const int l16  = lane & 15;

    // Stage W[hh] transposed into LDS as f16 (coalesced reads, packed b32 writes)
    {
        const float* wp = w + (size_t)hh * kFIN * kFOUT;
        const int n = tid & 63, q = tid >> 6;       // q in 0..3 covers k blocks
        for (int kk = 0; kk < 64; kk += 2) {
            const int k = q * 64 + kk;
            union { unsigned int u; _Float16 hx[2]; } pk;
            pk.hx[0] = (_Float16)wp[(size_t)k * kFOUT + n];
            pk.hx[1] = (_Float16)wp[(size_t)(k + 1) * kFOUT + n];
            *(unsigned int*)(w16T + n * kWPitch + k) = pk.u;
        }
    }
    __syncthreads();

    const int rowbase = blockIdx.y * 128 + wave * 16;
    const float* hrow = h + ((size_t)(b * kN + rowbase + l16)) * kFIN;

    v8f acc[4] = {};

#pragma unroll
    for (int ks = 0; ks < 8; ++ks) {
        // A fragment: two runs of 8 consecutive K -> 4x global_load_b128
        v16h a;
        {
            const float4 f0 = *(const float4*)(hrow + ks * 32 + half * 8);
            const float4 f1 = *(const float4*)(hrow + ks * 32 + half * 8 + 4);
            const float4 f2 = *(const float4*)(hrow + ks * 32 + 16 + half * 8);
            const float4 f3 = *(const float4*)(hrow + ks * 32 + 16 + half * 8 + 4);
            a[0]=(_Float16)f0.x; a[1]=(_Float16)f0.y; a[2]=(_Float16)f0.z; a[3]=(_Float16)f0.w;
            a[4]=(_Float16)f1.x; a[5]=(_Float16)f1.y; a[6]=(_Float16)f1.z; a[7]=(_Float16)f1.w;
            a[8]=(_Float16)f2.x; a[9]=(_Float16)f2.y; a[10]=(_Float16)f2.z; a[11]=(_Float16)f2.w;
            a[12]=(_Float16)f3.x; a[13]=(_Float16)f3.y; a[14]=(_Float16)f3.z; a[15]=(_Float16)f3.w;
        }
        // Load ALL B fragments first so ds waits stagger across the 4 WMMAs.
        union BU { v16h v; v8h hv[2]; } bu[4];
#pragma unroll
        for (int t = 0; t < 4; ++t) {
            const _Float16* bp = w16T + (t * 16 + l16) * kWPitch + ks * 32 + half * 8;
            bu[t].hv[0] = *(const v8h*)bp;
            bu[t].hv[1] = *(const v8h*)(bp + 16);
        }
#pragma unroll
        for (int t = 0; t < 4; ++t)
            acc[t] = __builtin_amdgcn_wmma_f32_16x16x32_f16(
                false, a, false, bu[t].v, (short)0, acc[t], false, false);
    }

    // Store transposed h_prime: 8 consecutive rows of one column = 1x b128 store
    {
        _Float16* hpT = hp16T + (size_t)bh * kN * kFOUT;
#pragma unroll
        for (int t = 0; t < 4; ++t) {
            v8h pk;
#pragma unroll
            for (int r = 0; r < 8; ++r) pk[r] = (_Float16)acc[t][r];
            *(v8h*)(hpT + (size_t)(t * 16 + l16) * kN + rowbase + half * 8) = pk;
        }
    }

    // tanh-dot attention scalars per row
    const float* asp = a_src + hh * kFOUT;
    const float* adp = a_dst + hh * kFOUT;
#pragma unroll
    for (int r = 0; r < 8; ++r) {
        const int row = rowbase + r + half * 8;
        float s_acc = 0.f, d_acc = 0.f;
#pragma unroll
        for (int t = 0; t < 4; ++t) {
            const float tv = fast_tanh(acc[t][r]);
            const int col = t * 16 + l16;
            s_acc += tv * asp[col];
            d_acc += tv * adp[col];
        }
#pragma unroll
        for (int o = 1; o < 16; o <<= 1) {
            s_acc += __shfl_xor(s_acc, o, 32);
            d_acc += __shfl_xor(d_acc, o, 32);
        }
        if (l16 == 0) {
            as_ws[bh * kN + row] = s_acc;
            ad_ws[bh * kN + row] = d_acc;
        }
    }
}

// ---------------------------------------------------------------------------
// Phase 2: masked leaky-relu softmax + P @ h_prime via WMMA.
// Log2-domain softmax (lrelu is scale-invariant for c>0), row max via a
// compare-only pass, denominator accumulated inside the WMMA A-build,
// double-buffered value staging, adjacency read directly from L2.
// ---------------------------------------------------------------------------
__global__ __launch_bounds__(256) void gat_phase2(
    const unsigned char* __restrict__ adj, const _Float16* __restrict__ hp16T,
    const float* __restrict__ as_ws, const float* __restrict__ ad_ws,
    const float* __restrict__ bias, float* __restrict__ out)
{
    __shared__ float d_all[kN];                       // log2e-scaled, 4 KB
    __shared__ float sS[128];                         // log2e-scaled
    __shared__ float mM[128];
    __shared__ float iD[128];
    __shared__ _Float16 hpcT[2][kFOUT * kVPitch];     // ping-pong value chunks, 34 KB

    const int bh   = blockIdx.x;
    const int b    = bh >> 3;
    const int tid  = threadIdx.x;
    const int wave = tid >> 5;
    const int lane = tid & 31;
    const int half = lane >> 4;
    const int l16  = lane & 15;
    const int rowblock = blockIdx.y * 128;

    for (int idx = tid; idx < kN; idx += 256) d_all[idx] = ad_ws[bh * kN + idx] * kLog2e;
    if (tid < 128) sS[tid] = as_ws[bh * kN + rowblock + tid] * kLog2e;
    __syncthreads();

    // ---- Pass A: row max only (no exp), dword adjacency reads ----
    for (int r = 0; r < 16; ++r) {
        const int rr = wave * 16 + r;
        const int i  = rowblock + rr;
        const float si = sS[rr];
        const unsigned int* arow32 =
            (const unsigned int*)(adj + ((size_t)(b * kN + i)) * kN);
        if (r + 1 < 16)
            __builtin_prefetch(adj + ((size_t)(b * kN + i + 1)) * kN + lane * 32, 0, 0);
        float m_loc = -1e30f;
        for (int jb = lane; jb < kN / 4; jb += 32) {
            const unsigned int m4 = arow32[jb];
            const int j0 = jb * 4;
            const float4 dv = *(const float4*)(d_all + j0);
            const float d4[4] = {dv.x, dv.y, dv.z, dv.w};
#pragma unroll
            for (int u = 0; u < 4; ++u) {
                float e = si + d4[u];
                e = fmaxf(e, 0.2f * e);               // leaky relu (scale-invariant)
                const bool valid = ((m4 >> (8 * u)) & 0xFFu) || (j0 + u == i);
                if (valid) m_loc = fmaxf(m_loc, e);
            }
        }
#pragma unroll
        for (int o = 16; o >= 1; o >>= 1)
            m_loc = fmaxf(m_loc, __shfl_xor(m_loc, o, 32));
        if (lane == 0) mM[rr] = m_loc;
    }
    __syncthreads();

    const float si_row = sS[wave * 16 + l16];
    const float mi_row = mM[wave * 16 + l16];
    const int   i_row  = rowblock + wave * 16 + l16;
    const int   rrA    = wave * 16 + l16;
    const unsigned char* arowp = adj + ((size_t)(b * kN + i_row)) * kN;

    // ---- Pass B: P @ V via WMMA, 8 chunks of 128 keys, ping-pong staging ----
    const unsigned int* vsrc =
        (const unsigned int*)(hp16T + (size_t)bh * kN * kFOUT);

    // stage chunk 0 into buffer 0
    for (int idx = tid; idx < kFOUT * 64; idx += 256) {
        const int n = idx >> 6, jd = idx & 63;
        ((unsigned int*)hpcT[0])[n * (kVPitch / 2) + jd] = vsrc[(size_t)n * (kN / 2) + jd];
    }
    __syncthreads();

    v8f acc[4] = {};
    float dsum = 0.f;                                 // per-row softmax denominator
    for (int c = 0; c < 8; ++c) {
        // stage next chunk into the other buffer while computing this one
        if (c + 1 < 8) {
            const unsigned int* s2 = vsrc + (c + 1) * 64;
            unsigned int* dst = (unsigned int*)hpcT[(c + 1) & 1];
            for (int idx = tid; idx < kFOUT * 64; idx += 256) {
                const int n = idx >> 6, jd = idx & 63;
                dst[n * (kVPitch / 2) + jd] = s2[(size_t)n * (kN / 2) + jd];
            }
        }
        const _Float16* vbuf = hpcT[c & 1];

#pragma unroll
        for (int ks = 0; ks < 4; ++ks) {
            // Mask bytes: two 8B-aligned runs read straight from L2.
            const uint2 am0 = *(const uint2*)(arowp + c * 128 + ks * 32 + half * 8);
            const uint2 am1 = *(const uint2*)(arowp + c * 128 + ks * 32 + 16 + half * 8);
            const unsigned int mw[4] = {am0.x, am0.y, am1.x, am1.y};

            // Probability A-fragment; denominator accumulated for free.
            v16h a;
#pragma unroll
            for (int v = 0; v < 8; ++v) {
                const int kb = ks * 32 + ((v & 3) << 1) + ((v >> 2) << 4) + (half << 3);
                const int j0 = c * 128 + kb;
                const float2 dd = *(const float2*)(d_all + j0);
                const unsigned int mword = mw[v >> 1];
                const unsigned int b0 = (mword >> (16 * (v & 1))) & 0xFFu;
                const unsigned int b1 = (mword >> (16 * (v & 1) + 8)) & 0xFFu;
                float e0 = si_row + dd.x; e0 = fmaxf(e0, 0.2f * e0);
                float e1 = si_row + dd.y; e1 = fmaxf(e1, 0.2f * e1);
                const bool ok0 = b0 || (j0 == i_row);
                const bool ok1 = b1 || (j0 + 1 == i_row);
                const float p0 = ok0 ? fast_exp2(e0 - mi_row) : 0.f;
                const float p1 = ok1 ? fast_exp2(e1 - mi_row) : 0.f;
                dsum += p0 + p1;
                a[2 * v]     = (_Float16)p0;
                a[2 * v + 1] = (_Float16)p1;
            }
            // Load ALL B fragments, then issue the 4 WMMAs.
            union BU { v16h v; v8h hv[2]; } bu[4];
#pragma unroll
            for (int t = 0; t < 4; ++t) {
                const _Float16* bp = vbuf + (t * 16 + l16) * kVPitch + ks * 32 + half * 8;
                bu[t].hv[0] = *(const v8h*)bp;
                bu[t].hv[1] = *(const v8h*)(bp + 16);
            }
#pragma unroll
            for (int t = 0; t < 4; ++t)
                acc[t] = __builtin_amdgcn_wmma_f32_16x16x32_f16(
                    false, a, false, bu[t].v, (short)0, acc[t], false, false);
        }
        __syncthreads();   // chunk c done; staged chunk c+1 visible
    }

    // Finish denominator: lane halves cover complementary K subsets.
    dsum += __shfl_xor(dsum, 16, 32);
    if (half == 0) iD[rrA] = 1.f / dsum;
    __syncthreads();

    // ---- Epilogue: scale by 1/denominator, add bias, store ----
#pragma unroll
    for (int r = 0; r < 8; ++r) {
        const int rr = wave * 16 + r + half * 8;
        const int i  = rowblock + rr;
        const float idv = iD[rr];
#pragma unroll
        for (int t = 0; t < 4; ++t) {
            const int col = t * 16 + l16;
            out[((size_t)bh * kN + i) * kFOUT + col] = acc[t][r] * idv + bias[col];
        }
    }
}

// ---------------------------------------------------------------------------
extern "C" void kernel_launch(void* const* d_in, const int* in_sizes, int n_in,
                              void* d_out, int out_size, void* d_ws, size_t ws_size,
                              hipStream_t stream) {
    const float*         h     = (const float*)d_in[0];
    const unsigned char* adj   = (const unsigned char*)d_in[1];   // bool -> 1 byte
    const float*         w     = (const float*)d_in[2];
    const float*         a_src = (const float*)d_in[3];
    const float*         a_dst = (const float*)d_in[4];
    const float*         bias  = (const float*)d_in[5];
    float*               out   = (float*)d_out;

    // Workspace: hp16T (8 MB, transposed) | attn_src (256 KB) | attn_dst (256 KB)
    _Float16* hp16T = (_Float16*)d_ws;
    float*    as_ws = (float*)((char*)d_ws + (size_t)kBS * kNH * kN * kFOUT * 2);
    float*    ad_ws = as_ws + (size_t)kBS * kNH * kN;

    dim3 blk(256);
    dim3 grid(kBS * kNH, kN / 128);
    gat_phase1<<<grid, blk, 0, stream>>>(h, w, a_src, a_dst, hp16T, as_ws, ad_ws);
    gat_phase2<<<grid, blk, 0, stream>>>(adj, hp16T, as_ws, ad_ws, bias, out);
}
